// LoadCheckpoint_70205535420771
// MI455X (gfx1250) — matrix operand order repaired and checked by reading, not compiled, not run on hardware
//
#include <hip/hip_runtime.h>
#include <hip/hip_bf16.h>

typedef __attribute__((ext_vector_type(16))) __bf16 v16bf;
typedef __attribute__((ext_vector_type(8)))  __bf16 v8bf;
typedef __attribute__((ext_vector_type(8)))  float  v8f;
typedef __attribute__((ext_vector_type(4)))  unsigned int v4u;
typedef __attribute__((ext_vector_type(8)))  int v8i;
typedef __attribute__((ext_vector_type(4)))  int v4i;

#define BB 16
#define SS 512
#define DD 384
#define HH 12
#define DH 32
#define LL 6
#define FF 1536
#define BS (BB*SS)

#define KC 128                 // K-chunk staged per TDM transfer
#define LPAD 8                 // +16B per LDS row -> bank skew of 4
#define LROW (KC + LPAD)       // LDS row stride in bf16 elements (272 B)
#define LDSBUF (64 * LROW * 2) // bytes per LDS buffer (17408)

__device__ __forceinline__ float sigmoid_f(float x) { return 1.0f / (1.0f + expf(-x)); }
__device__ __forceinline__ float gelu_f(float x) { return 0.5f * x * (1.0f + erff(x * 0.70710678118654752f)); }

// ---------------------------------------------------------------------------
// TDM: DMA a 64-row x KC-col bf16 tile (row stride K elems) from global into
// LDS at byte offset lds_off, with 16B padding inserted after every 256B row
// (pad_interval=64 dwords -> code 5, pad_amount=4 dwords -> code 3).
// D# built per CDNA5 ISA 8.3/8.4 (group0: count/lds/global/type=2;
// group1: data_size=2B, pad cfg, tensor_dim0=K, tensor_dim1=N,
// tile_dim0=KC, tile_dim1=64, tensor_dim0_stride=K).
// ---------------------------------------------------------------------------
__device__ __forceinline__ void tdm_load_tile(const __bf16* gptr, unsigned lds_off,
                                              int K, int Nrows)
{
    unsigned long long ga = (unsigned long long)(uintptr_t)gptr;
    v4u g0;
    g0[0] = 1u;                                                   // count=1
    g0[1] = lds_off;                                              // lds_addr
    g0[2] = (unsigned)ga;                                         // global_addr lo
    g0[3] = (unsigned)((ga >> 32) & 0x1FFFFFFu) | (2u << 30);     // addr hi | type=2

    unsigned long long q0 = (1ull << 16)   // data_size = 2 bytes
                          | (1ull << 20)   // pad_enable
                          | (5ull << 22)   // pad_interval: 64 dwords
                          | (3ull << 25)   // pad_amount: 4 dwords
                          | ((unsigned long long)((unsigned)K & 0xFFFFu) << 48); // tensor_dim0 lo
    unsigned long long q1 = (unsigned long long)((unsigned)K >> 16)              // tensor_dim0 hi
                          | ((unsigned long long)(unsigned)Nrows << 16)          // tensor_dim1
                          | ((unsigned long long)(KC & 0xFFFF) << 48);           // tile_dim0
    unsigned long long q2 = 64ull                                                // tile_dim1
                          | ((unsigned long long)(unsigned)K << 32);             // dim0_stride lo32
    unsigned long long q3 = 0ull;                                                // stride hi / dim1_stride

    v8i g1;
    g1[0] = (int)(unsigned)q0; g1[1] = (int)(q0 >> 32);
    g1[2] = (int)(unsigned)q1; g1[3] = (int)(q1 >> 32);
    g1[4] = (int)(unsigned)q2; g1[5] = (int)(q2 >> 32);
    g1[6] = (int)(unsigned)q3; g1[7] = (int)(q3 >> 32);
    v4i z4 = {0, 0, 0, 0};
#if defined(__clang_major__) && (__clang_major__ >= 23)
    v8i z8 = {0, 0, 0, 0, 0, 0, 0, 0};
    __builtin_amdgcn_tensor_load_to_lds(g0, g1, z4, z4, z8, 0);
#else
    __builtin_amdgcn_tensor_load_to_lds(g0, g1, z4, z4, 0);
#endif
}

// ---------------------------------------------------------------------------
// Embedding gather: x[t,d] = word_emb[ids[t],d] + pos_emb[s,d] + type_emb[0,d]
// ---------------------------------------------------------------------------
__global__ __launch_bounds__(256) void embed_kernel(
    const int* __restrict__ ids, const float* __restrict__ wemb,
    const float* __restrict__ pemb, const float* __restrict__ temb,
    float* __restrict__ x)
{
    int idx = blockIdx.x * 256 + threadIdx.x;
    if (idx >= BS * DD) return;
    int t = idx / DD, d = idx - t * DD;
    int s = t & (SS - 1);
    x[idx] = wemb[(size_t)ids[t] * DD + d] + pemb[(size_t)s * DD + d] + temb[d];
}

// ---------------------------------------------------------------------------
// Fused (optional residual add) + LayerNorm. One wave per token (D=384 = 12x32).
// ---------------------------------------------------------------------------
__global__ __launch_bounds__(256) void add_ln_kernel(
    float* __restrict__ x, const float* __restrict__ res,
    const float* __restrict__ g, const float* __restrict__ bta)
{
    int wave = threadIdx.x >> 5, lane = threadIdx.x & 31;
    int t = blockIdx.x * 8 + wave;
    if (t >= BS) return;
    const float* xr = x + (size_t)t * DD;
    const float* rr = res ? res + (size_t)t * DD : nullptr;
    float v[12];
    float s = 0.f;
#pragma unroll
    for (int i = 0; i < 12; ++i) {
        float xv = xr[lane + 32 * i];
        if (rr) xv += rr[lane + 32 * i];
        v[i] = xv; s += xv;
    }
#pragma unroll
    for (int o = 16; o >= 1; o >>= 1) s += __shfl_xor(s, o, 32);
    float mean = s * (1.0f / DD);
    float vs = 0.f;
#pragma unroll
    for (int i = 0; i < 12; ++i) { float d = v[i] - mean; vs += d * d; }
#pragma unroll
    for (int o = 16; o >= 1; o >>= 1) vs += __shfl_xor(vs, o, 32);
    float inv = rsqrtf(vs * (1.0f / DD) + 1e-12f);
    float* xw = x + (size_t)t * DD;
#pragma unroll
    for (int i = 0; i < 12; ++i) {
        int d = lane + 32 * i;
        xw[d] = (v[i] - mean) * inv * g[d] + bta[d];
    }
}

// ---------------------------------------------------------------------------
// f32 -> bf16 convert (activations)
// ---------------------------------------------------------------------------
__global__ __launch_bounds__(256) void acvt_kernel(
    const float* __restrict__ s, __bf16* __restrict__ d, int n)
{
    int i = blockIdx.x * 256 + threadIdx.x;
    if (i < n) d[i] = (__bf16)s[i];
}

// ---------------------------------------------------------------------------
// Weight transpose-convert: src f32 [K,N] -> dst bf16 [N,K]. LDS 32x33 tile.
// ---------------------------------------------------------------------------
__global__ __launch_bounds__(256) void wcvt_kernel(
    const float* __restrict__ src, __bf16* __restrict__ dst, int K, int N)
{
    __shared__ float tile[32][33];
    int kx = blockIdx.x * 32, ny = blockIdx.y * 32;
    int tx = threadIdx.x & 31, ty = threadIdx.x >> 5;  // ty in [0,8)
#pragma unroll
    for (int p = 0; p < 4; ++p)
        tile[ty + p * 8][tx] = src[(size_t)(kx + ty + p * 8) * N + (ny + tx)];
    __syncthreads();
#pragma unroll
    for (int p = 0; p < 4; ++p)
        dst[(size_t)(ny + ty + p * 8) * K + (kx + tx)] = (__bf16)tile[tx][ty + p * 8];
}

// ---------------------------------------------------------------------------
// WMMA bf16 GEMM: C[M,N] = A[M,K] @ Wt[N,K]^T + bias.
// Block = 128 threads (4 waves), block tile 64(M) x 64(N); each wave owns a
// 16x64 strip. The 64xK weight tile is shared by all 4 waves, so it is staged
// through LDS by the Tensor Data Mover in KC-wide double-buffered chunks
// (wave 0 issues tensor_load_to_lds; handoff via s_wait_tensorcnt + barrier).
// A fragments stream from global (L2-resident). EPI=0: f32 store.
// EPI=1: exact-erf GELU then bf16 store.
// ---------------------------------------------------------------------------
template <int EPI>
__global__ __launch_bounds__(128) void gemm_bf16_kernel(
    const __bf16* __restrict__ A, const __bf16* __restrict__ Wt,
    const float* __restrict__ bias, void* __restrict__ Cout,
    int M, int N, int K)
{
    __shared__ __bf16 lds[2][64][LROW];
    const int wave = threadIdx.x >> 5;
    const int lane = threadIdx.x & 31;
    const int m0 = (blockIdx.x * 4 + wave) * 16;
    const int n0 = blockIdx.y * 64;
    const int al = lane & 15;
    const int kb = (lane < 16) ? 0 : 16;   // A K-half select
    const int kB = (lane < 16) ? 0 : 16;   // B K-half select
    const int nchunks = K / KC;

    // Prologue: stage chunk 0 of the weight tile.
    if (threadIdx.x < 32) {
        tdm_load_tile(Wt + (size_t)n0 * K, 0u, K, N);
        __builtin_amdgcn_s_wait_tensorcnt(0);
    }
    __syncthreads();

    v8f acc[4] = {};
    const __bf16* arow = A + (size_t)(m0 + al) * K;

    for (int c = 0; c < nchunks; ++c) {
        const int cb = c & 1;
        if (c + 1 < nchunks) {
            if (threadIdx.x < 32)
                tdm_load_tile(Wt + (size_t)n0 * K + (c + 1) * KC,
                              (unsigned)((cb ^ 1) * LDSBUF), K, N);
            __builtin_prefetch(arow + (c + 1) * KC, 0, 0);  // global_prefetch_b8
        }
#pragma unroll
        for (int ks = 0; ks < KC; ks += 32) {
            // A fragment (16x32 bf16, split-K-half per lane group)
            const __bf16* ap = arow + c * KC + ks;
            v8bf alo = *(const v8bf*)(ap + kb);
            v8bf ahi = *(const v8bf*)(ap + kb + 8);
            v16bf a;
#pragma unroll
            for (int i = 0; i < 8; ++i) { a[i] = alo[i]; a[8 + i] = ahi[i]; }
            // Four B fragments from LDS (contiguous 32B per lane, bank-skewed)
            v16bf b0 = *(const v16bf*)&lds[cb][0 * 16 + al][ks + kB];
            v16bf b1 = *(const v16bf*)&lds[cb][1 * 16 + al][ks + kB];
            v16bf b2 = *(const v16bf*)&lds[cb][2 * 16 + al][ks + kB];
            v16bf b3 = *(const v16bf*)&lds[cb][3 * 16 + al][ks + kB];
            acc[0] = __builtin_amdgcn_wmma_f32_16x16x32_bf16(false, a, false, b0, (short)0, acc[0], false, false);
            acc[1] = __builtin_amdgcn_wmma_f32_16x16x32_bf16(false, a, false, b1, (short)0, acc[1], false, false);
            acc[2] = __builtin_amdgcn_wmma_f32_16x16x32_bf16(false, a, false, b2, (short)0, acc[2], false, false);
            acc[3] = __builtin_amdgcn_wmma_f32_16x16x32_bf16(false, a, false, b3, (short)0, acc[3], false, false);
        }
        if (threadIdx.x < 32) __builtin_amdgcn_s_wait_tensorcnt(0);
        __syncthreads();
    }

    const int cn = lane & 15;
    const int mhi = (lane >> 4) * 8;
#pragma unroll
    for (int nt = 0; nt < 4; ++nt) {
        int n = n0 + nt * 16 + cn;
        float bv = bias[n];
#pragma unroll
        for (int r = 0; r < 8; ++r) {
            int m = m0 + mhi + r;
            float val = acc[nt][r] + bv;
            if (EPI == 1) {
                ((__bf16*)Cout)[(size_t)m * N + n] = (__bf16)gelu_f(val);
            } else {
                ((float*)Cout)[(size_t)m * N + n] = val;
            }
        }
    }
}

// ---------------------------------------------------------------------------
// Beta gate: beta[t,h] = mask[t] * sigmoid(x[t,:] @ Wb[:,h] + bb[h])
// ---------------------------------------------------------------------------
__global__ __launch_bounds__(128) void beta_kernel(
    const float* __restrict__ x, const float* __restrict__ Wb,
    const float* __restrict__ bb, const int* __restrict__ amask,
    float* __restrict__ beta)
{
    int idx = blockIdx.x * 128 + threadIdx.x;
    if (idx >= BS * HH) return;
    int t = idx / HH, h = idx - t * HH;
    float acc = bb[h];
    const float* xr = x + (size_t)t * DD;
    for (int j = 0; j < DD; ++j) acc = fmaf(xr[j], Wb[(size_t)j * HH + h], acc);
    beta[idx] = (float)amask[t] * sigmoid_f(acc);
}

// ---------------------------------------------------------------------------
// L2-normalize q and k per head row (DH=32 == wave width: lane-per-element)
// ---------------------------------------------------------------------------
__global__ __launch_bounds__(256) void qknorm_kernel(
    float* __restrict__ q, float* __restrict__ k)
{
    int e = blockIdx.x * 256 + threadIdx.x;  // aligned: 32 lanes = one head row
    float qv = q[e], kv = k[e];
    float sq = qv * qv, sk = kv * kv;
#pragma unroll
    for (int o = 16; o >= 1; o >>= 1) { sq += __shfl_xor(sq, o, 32); sk += __shfl_xor(sk, o, 32); }
    q[e] = qv / (sqrtf(sq) + 1e-6f);
    k[e] = kv / (sqrtf(sk) + 1e-6f);
}

// ---------------------------------------------------------------------------
// DeltaNet dual-state scan. Block = 64 threads: wave 0 = fast state,
// wave 1 = slow state. Lane v owns state column S[k][v], k in 32 VGPRs.
// Broadcasts via __shfl; states mixed through 32 floats of LDS.
// ---------------------------------------------------------------------------
__global__ __launch_bounds__(64) void scan_kernel(
    const float* __restrict__ q, const float* __restrict__ k,
    const float* __restrict__ v, const float* __restrict__ beta,
    const float* __restrict__ dfast, const float* __restrict__ dslow,
    const float* __restrict__ alpha, float* __restrict__ out)
{
    __shared__ float so[32];
    int bh = blockIdx.x;
    int b = bh / HH, h = bh - b * HH;
    int wave = threadIdx.x >> 5, lane = threadIdx.x & 31;
    float g = sigmoid_f(wave ? dslow[h] : dfast[h]);
    float aw = sigmoid_f(alpha[h]);
    float w = wave ? (1.0f - aw) : aw;

    float S[32];
#pragma unroll
    for (int j = 0; j < 32; ++j) S[j] = 0.f;

    for (int t = 0; t < SS; ++t) {
        size_t base = ((size_t)(b * SS + t) * HH + h) * DH;
        float kt = k[base + lane];
        float vt = v[base + lane];
        float qt = q[base + lane];
        float bt = beta[(size_t)(b * SS + t) * HH + h];
        float kbc[32];
#pragma unroll
        for (int j = 0; j < 32; ++j) kbc[j] = __shfl(kt, j, 32);
        float pred = 0.f;
#pragma unroll
        for (int j = 0; j < 32; ++j) { S[j] *= g; pred = fmaf(S[j], kbc[j], pred); }
        float dl = bt * (vt - pred);
        float o = 0.f;
#pragma unroll
        for (int j = 0; j < 32; ++j) {
            S[j] = fmaf(kbc[j], dl, S[j]);
            o = fmaf(S[j], __shfl(qt, j, 32), o);
        }
        o *= w;
        if (wave) so[lane] = o;
        __syncthreads();
        if (!wave) out[base + lane] = o + so[lane];
        __syncthreads();
    }
}

// ---------------------------------------------------------------------------
// Masked mean pool + L2 normalize. One block per batch, 384 threads.
// ---------------------------------------------------------------------------
__global__ __launch_bounds__(384) void pool_kernel(
    const float* __restrict__ x, const int* __restrict__ amask,
    float* __restrict__ out)
{
    __shared__ float red[384];
    int b = blockIdx.x, d = threadIdx.x;
    float s = 0.f, cnt = 0.f;
    for (int t = 0; t < SS; ++t) {
        float m = (float)amask[b * SS + t];
        s = fmaf(m, x[((size_t)(b * SS + t)) * DD + d], s);
        cnt += m;
    }
    float p = s / fmaxf(cnt, 1e-9f);
    red[d] = p * p;
    __syncthreads();
    if (d < 128) red[d] += red[d + 128] + red[d + 256];
    __syncthreads();
    for (int st = 64; st >= 1; st >>= 1) {
        if (d < st) red[d] += red[d + st];
        __syncthreads();
    }
    float nrm = fmaxf(sqrtf(red[0]), 1e-12f);
    out[(size_t)b * DD + d] = p / nrm;
}

// ---------------------------------------------------------------------------
extern "C" void kernel_launch(void* const* d_in, const int* in_sizes, int n_in,
                              void* d_out, int out_size, void* d_ws, size_t ws_size,
                              hipStream_t stream)
{
    (void)in_sizes; (void)n_in; (void)out_size; (void)ws_size;
    const int*   ids    = (const int*)d_in[0];
    const int*   amask  = (const int*)d_in[1];
    const float* wemb   = (const float*)d_in[2];
    const float* pemb   = (const float*)d_in[3];
    const float* temb   = (const float*)d_in[4];
    const float* elng   = (const float*)d_in[5];
    const float* elnb   = (const float*)d_in[6];
    const float* Wq     = (const float*)d_in[7];
    const float* bq     = (const float*)d_in[8];
    const float* Wk     = (const float*)d_in[9];
    const float* bk     = (const float*)d_in[10];
    const float* Wv     = (const float*)d_in[11];
    const float* bv     = (const float*)d_in[12];
    const float* Wbeta  = (const float*)d_in[13];
    const float* bbeta  = (const float*)d_in[14];
    const float* dfast  = (const float*)d_in[15];
    const float* dslow  = (const float*)d_in[16];
    const float* alpha  = (const float*)d_in[17];
    const float* Wo     = (const float*)d_in[18];
    const float* bo     = (const float*)d_in[19];
    const float* ln1g   = (const float*)d_in[20];
    const float* ln1b   = (const float*)d_in[21];
    const float* W1     = (const float*)d_in[22];
    const float* b1     = (const float*)d_in[23];
    const float* W2     = (const float*)d_in[24];
    const float* b2     = (const float*)d_in[25];
    const float* ln2g   = (const float*)d_in[26];
    const float* ln2b   = (const float*)d_in[27];

    // Workspace layout (bytes)
    const size_t XSZ = (size_t)BS * DD * sizeof(float);      // 12,582,912
    char* wsb = (char*)d_ws;
    float*  x    = (float*)(wsb);
    float*  qb   = (float*)(wsb + 1 * XSZ);
    float*  kb   = (float*)(wsb + 2 * XSZ);
    float*  vb   = (float*)(wsb + 3 * XSZ);
    float*  t4   = (float*)(wsb + 4 * XSZ);
    float*  betab= (float*)(wsb + 5 * XSZ);                  // BS*H f32
    __bf16* xb   = (__bf16*)(wsb + 5 * XSZ + (size_t)BS * HH * 4);   // BS*D bf16
    __bf16* hb   = (__bf16*)((char*)xb + (size_t)BS * DD * 2);       // BS*FF bf16
    __bf16* wb   = (__bf16*)((char*)hb + (size_t)BS * FF * 2);       // FF*D bf16

    // Embedding + LN
    embed_kernel<<<(BS * DD + 255) / 256, 256, 0, stream>>>(ids, wemb, pemb, temb, x);
    add_ln_kernel<<<BS / 8, 256, 0, stream>>>(x, nullptr, elng, elnb);

    const dim3 gP(BS / 64, DD / 64);   // N=384 GEMMs
    const dim3 gF1(BS / 64, FF / 64);  // N=1536 GEMM
    const dim3 wDD(DD / 32, DD / 32);
    const dim3 wDF(DD / 32, FF / 32);
    const dim3 wFD(FF / 32, DD / 32);

    for (int i = 0; i < LL; ++i) {
        const size_t oDD = (size_t)i * DD * DD;
        // x -> bf16 once, used for q/k/v projections
        acvt_kernel<<<(BS * DD + 255) / 256, 256, 0, stream>>>(x, xb, BS * DD);
        // Q
        wcvt_kernel<<<wDD, 256, 0, stream>>>(Wq + oDD, wb, DD, DD);
        gemm_bf16_kernel<0><<<gP, 128, 0, stream>>>(xb, wb, bq + (size_t)i * DD, qb, BS, DD, DD);
        // K
        wcvt_kernel<<<wDD, 256, 0, stream>>>(Wk + oDD, wb, DD, DD);
        gemm_bf16_kernel<0><<<gP, 128, 0, stream>>>(xb, wb, bk + (size_t)i * DD, kb, BS, DD, DD);
        // V
        wcvt_kernel<<<wDD, 256, 0, stream>>>(Wv + oDD, wb, DD, DD);
        gemm_bf16_kernel<0><<<gP, 128, 0, stream>>>(xb, wb, bv + (size_t)i * DD, vb, BS, DD, DD);
        // beta gate (small N=12: VALU kernel, f32 precision)
        beta_kernel<<<(BS * HH + 127) / 128, 128, 0, stream>>>(
            x, Wbeta + (size_t)i * DD * HH, bbeta + (size_t)i * HH, amask, betab);
        // q/k row normalization
        qknorm_kernel<<<BS * DD / 256, 256, 0, stream>>>(qb, kb);
        // dual-state delta scan -> t4
        scan_kernel<<<BB * HH, 64, 0, stream>>>(
            qb, kb, vb, betab,
            dfast + (size_t)i * HH, dslow + (size_t)i * HH, alpha + (size_t)i * HH, t4);
        // output projection: t4 -> qb (reuse)
        acvt_kernel<<<(BS * DD + 255) / 256, 256, 0, stream>>>(t4, xb, BS * DD);
        wcvt_kernel<<<wDD, 256, 0, stream>>>(Wo + oDD, wb, DD, DD);
        gemm_bf16_kernel<0><<<gP, 128, 0, stream>>>(xb, wb, bo + (size_t)i * DD, qb, BS, DD, DD);
        // residual + LN1
        add_ln_kernel<<<BS / 8, 256, 0, stream>>>(x, qb, ln1g + (size_t)i * DD, ln1b + (size_t)i * DD);
        // FFN1: x @ W1 + b1, GELU, store bf16 -> hb
        acvt_kernel<<<(BS * DD + 255) / 256, 256, 0, stream>>>(x, xb, BS * DD);
        wcvt_kernel<<<wDF, 256, 0, stream>>>(W1 + (size_t)i * DD * FF, wb, DD, FF);
        gemm_bf16_kernel<1><<<gF1, 128, 0, stream>>>(xb, wb, b1 + (size_t)i * FF, hb, BS, FF, DD);
        // FFN2: hb @ W2 + b2 -> qb
        wcvt_kernel<<<wFD, 256, 0, stream>>>(W2 + (size_t)i * FF * DD, wb, FF, DD);
        gemm_bf16_kernel<0><<<gP, 128, 0, stream>>>(hb, wb, b2 + (size_t)i * DD, qb, BS, DD, FF);
        // residual + LN2
        add_ln_kernel<<<BS / 8, 256, 0, stream>>>(x, qb, ln2g + (size_t)i * DD, ln2b + (size_t)i * DD);
    }

    pool_kernel<<<BB, DD, 0, stream>>>(x, amask, (float*)d_out);
}
